// BilinearDecoder_35871566856222
// MI455X (gfx1250) — matrix operand order repaired and checked by reading, not compiled, not run on hardware
//
#include <hip/hip_runtime.h>
#include <hip/hip_bf16.h>

typedef __bf16 bf16_t;
typedef unsigned short u16;
typedef __attribute__((ext_vector_type(8)))  __bf16 v8bf;
typedef __attribute__((ext_vector_type(16))) __bf16 v16bf;
typedef __attribute__((ext_vector_type(8)))  float  v8f;

#define FDIM 512
#define NENT 4096
#define CELLS 39
#define PAIRS 8192

// ---------- bf16 helpers (round-to-nearest-even) ----------
__device__ __forceinline__ u16 f32_to_bf16(float f) {
  unsigned int u = __float_as_uint(f);
  unsigned int r = 0x7FFFu + ((u >> 16) & 1u);
  return (u16)((u + r) >> 16);
}
__device__ __forceinline__ float bf16_to_f32(u16 h) {
  return __uint_as_float(((unsigned int)h) << 16);
}

// ---------- kernel 1: split embedding into bf16 hi/lo ----------
__global__ __launch_bounds__(256) void split_emb_kernel(
    const float* __restrict__ emb, u16* __restrict__ hi, u16* __restrict__ lo) {
  int i = blockIdx.x * 256 + threadIdx.x;        // over NENT*FDIM
  float v = emb[i];
  u16 h = f32_to_bf16(v);
  hi[i] = h;
  lo[i] = f32_to_bf16(v - bf16_to_f32(h));
}

// ---------- kernel 2: build Wc^T (N-major) with hi/lo split ----------
// Wc[n,k] = wl[n]*W[n,k]*wl[k]; stored transposed: WcT[k*FDIM + n]
__global__ __launch_bounds__(256) void build_wct_kernel(
    const float* __restrict__ W, const float* __restrict__ wl_row,
    u16* __restrict__ bhi, u16* __restrict__ blo) {
  int idx = blockIdx.x * 256 + threadIdx.x;      // idx = k*FDIM + n (coalesced store)
  int k = idx >> 9;
  int n = idx & (FDIM - 1);
  float v = wl_row[n] * W[n * FDIM + k] * wl_row[k];
  u16 h = f32_to_bf16(v);
  bhi[idx] = h;
  blo[idx] = f32_to_bf16(v - bf16_to_f32(h));
}

// ---------- WMMA fragment loads (wave32 layouts per CDNA5 ISA 7.12.2) ----------
// A (16x32 bf16, row-major src): lane row = lane&15; lanes<16 hold K {0..7,16..23},
// lanes>=16 hold K {8..15,24..31}. Two contiguous 16-byte chunks.
__device__ __forceinline__ v16bf load_frag_a(const u16* __restrict__ tile, int lane) {
  int row  = lane & 15;
  int koff = (lane >> 4) << 3;                   // 0 or 8
  const v8bf* p = (const v8bf*)(tile + row * FDIM + koff);
  v8bf c0 = p[0];                                // K = koff .. koff+7
  v8bf c1 = p[2];                                // K = koff+16 .. koff+23
  return __builtin_shufflevector(c0, c1, 0,1,2,3,4,5,6,7,8,9,10,11,12,13,14,15);
}
// B (32x16 bf16) from N-major WcT: lane col N = lane&15; lanes<16 hold K {0..7,16..23},
// lanes>=16 hold K {8..15,24..31}. Two contiguous 16-byte chunks.
__device__ __forceinline__ v16bf load_frag_b(const u16* __restrict__ tile, int lane) {
  int col  = lane & 15;
  int koff = (lane >> 4) << 3;                   // 0 or 8
  const v8bf* p = (const v8bf*)(tile + col * FDIM + koff);
  v8bf c0 = p[0];
  v8bf c1 = p[2];
  return __builtin_shufflevector(c0, c1, 0,1,2,3,4,5,6,7,8,9,10,11,12,13,14,15);
}

// ---------- kernel 3: F = E @ Wc via bf16x3 WMMA ----------
// E: NENT x FDIM (K = n), WcT: FDIM(N=k) x FDIM(K=n), F: NENT x FDIM fp32.
// Block = 256 threads = 8 waves in 4(M) x 2(N); wave tile 32x32 (2x2 WMMA frags).
__global__ __launch_bounds__(256) void gemm_bf16x3_kernel(
    const u16* __restrict__ Ehi, const u16* __restrict__ Elo,
    const u16* __restrict__ Bhi, const u16* __restrict__ Blo,
    float* __restrict__ F) {
  const int lane = threadIdx.x & 31;
  const int wave = threadIdx.x >> 5;
  const int wm = wave & 3;
  const int wn = wave >> 2;
  const int m0 = blockIdx.x * 128 + wm * 32;
  const int n0 = blockIdx.y * 64 + wn * 32;

  v8f acc[2][2] = {};
  for (int kb = 0; kb < FDIM; kb += 32) {
    v16bf ah[2], al[2], bh[2], bl[2];
#pragma unroll
    for (int i = 0; i < 2; ++i) {
      ah[i] = load_frag_a(Ehi + (m0 + 16 * i) * FDIM + kb, lane);
      al[i] = load_frag_a(Elo + (m0 + 16 * i) * FDIM + kb, lane);
      bh[i] = load_frag_b(Bhi + (n0 + 16 * i) * FDIM + kb, lane);
      bl[i] = load_frag_b(Blo + (n0 + 16 * i) * FDIM + kb, lane);
    }
#pragma unroll
    for (int mi = 0; mi < 2; ++mi)
#pragma unroll
      for (int ni = 0; ni < 2; ++ni) {
        // hi*hi + hi*lo + lo*hi  (lo*lo term negligible at fp32 target accuracy)
        acc[mi][ni] = __builtin_amdgcn_wmma_f32_16x16x32_bf16(
            false, ah[mi], false, bh[ni], (short)0, acc[mi][ni], false, false);
        acc[mi][ni] = __builtin_amdgcn_wmma_f32_16x16x32_bf16(
            false, ah[mi], false, bl[ni], (short)0, acc[mi][ni], false, false);
        acc[mi][ni] = __builtin_amdgcn_wmma_f32_16x16x32_bf16(
            false, al[mi], false, bh[ni], (short)0, acc[mi][ni], false, false);
      }
  }
  // D layout: element r -> row m0+16*mi + r + (lane>=16 ? 8 : 0), col n0+16*ni+(lane&15)
  const int rofs = (lane >> 4) << 3;
  const int cofs = lane & 15;
#pragma unroll
  for (int mi = 0; mi < 2; ++mi)
#pragma unroll
    for (int ni = 0; ni < 2; ++ni) {
      int col   = n0 + 16 * ni + cofs;
      int rbase = m0 + 16 * mi + rofs;
#pragma unroll
      for (int r = 0; r < 8; ++r)
        F[(rbase + r) * FDIM + col] = acc[mi][ni][r];
    }
}

// ---------- kernel 4: out[p] = dot(F[i0], emb[i1]) ; one wave per pair ----------
__global__ __launch_bounds__(256) void gather_dot_kernel(
    const float* __restrict__ F, const float* __restrict__ emb,
    const int* __restrict__ idx_cell, float* __restrict__ out_cell) {
  int pair = blockIdx.x * 8 + (threadIdx.x >> 5);
  int lane = threadIdx.x & 31;
  int i0 = idx_cell[2 * pair];
  int i1 = idx_cell[2 * pair + 1];
  const float4* fr = (const float4*)(F + (size_t)i0 * FDIM);
  const float4* er = (const float4*)(emb + (size_t)i1 * FDIM);
  float s = 0.f;
#pragma unroll
  for (int j = 0; j < FDIM / (32 * 4); ++j) {   // 4 iters of float4 per lane
    float4 a = fr[lane + 32 * j];
    float4 b = er[lane + 32 * j];
    s += a.x * b.x + a.y * b.y + a.z * b.z + a.w * b.w;
  }
#pragma unroll
  for (int off = 16; off > 0; off >>= 1) s += __shfl_down(s, off, 32);
  if (lane == 0) out_cell[pair] = s;
}

extern "C" void kernel_launch(void* const* d_in, const int* in_sizes, int n_in,
                              void* d_out, int out_size, void* d_ws, size_t ws_size,
                              hipStream_t stream) {
  (void)in_sizes; (void)n_in; (void)out_size; (void)ws_size;
  const float* emb  = (const float*)d_in[0];   // (4096, 512) f32
  const int*   idx  = (const int*)d_in[1];     // (39, 8192, 2) i32
  const float* Wg   = (const float*)d_in[2];   // (512, 512) f32
  const float* wl   = (const float*)d_in[3];   // (39, 512) f32
  float* out = (float*)d_out;                  // (39, 8192) f32

  // Workspace layout (~17 MB):
  char* ws = (char*)d_ws;
  u16*   Ehi = (u16*)(ws);                               // 4 MB
  u16*   Elo = (u16*)(ws + (4u << 20));                  // 4 MB
  u16*   Bhi = (u16*)(ws + (8u << 20));                  // 512 KB
  u16*   Blo = (u16*)(ws + (8u << 20) + (512u << 10));   // 512 KB
  float* F   = (float*)(ws + (9u << 20));                // 8 MB

  split_emb_kernel<<<(NENT * FDIM) / 256, 256, 0, stream>>>(emb, Ehi, Elo);

  for (int c = 0; c < CELLS; ++c) {
    build_wct_kernel<<<(FDIM * FDIM) / 256, 256, 0, stream>>>(
        Wg, wl + (size_t)c * FDIM, Bhi, Blo);
    dim3 grid(NENT / 128, FDIM / 64);
    gemm_bf16x3_kernel<<<grid, 256, 0, stream>>>(Ehi, Elo, Bhi, Blo, F);
    gather_dot_kernel<<<PAIRS / 8, 256, 0, stream>>>(
        F, emb, idx + (size_t)c * PAIRS * 2, out + (size_t)c * PAIRS);
  }
}
